// diagonal_LSTM_66073776881699
// MI455X (gfx1250) — matrix operand order repaired and checked by reading, not compiled, not run on hardware
//
#include <hip/hip_runtime.h>

// ---------------------------------------------------------------------------
// Diagonal LSTM for MI455X (gfx1250, wave32, WMMA bf16) — v2
//   B=32, C=128, H=32, W=32, O=4C=512, T=63
// One 1024-thread workgroup per batch. Gate-interleaved weight packing:
//   packed row p = ch*4 + gate  (orig row = gate*128 + ch)
// so each lane's WMMA accumulators hold complete (o,f,i,g) quadruples for
// 2 channels -> gate math entirely in registers, no z LDS buffer, one
// barrier per step (double-buffered x/h staging).
// [W0|W1] A-fragments live in LDS (256KB); W_is A-fragments in registers.
// ---------------------------------------------------------------------------

typedef __attribute__((ext_vector_type(16))) __bf16 v16bf;
typedef __attribute__((ext_vector_type(8)))  float  v8f;

#define B_  32
#define C_  128
#define H_  32
#define W_  32
#define O_  512
#define T_  63

// fp32 -> bf16 (round to nearest even), stored as u16
__device__ __forceinline__ unsigned short f2bf(float f) {
    unsigned u = __builtin_bit_cast(unsigned, f);
    unsigned r = u + 0x7FFFu + ((u >> 16) & 1u);
    return (unsigned short)(r >> 16);
}

// ---------------------------------------------------------------------------
// Prep: pack W_is and [W0|W1] into bf16 A-fragment order with gate-interleaved
// row permutation, and fold the two biases.
// A 16x32 bf16 frag (contiguous k order): lane l -> M = l&15;
//   elems 0..15 -> K = kb+0..15, kb = (l>>4)*16.
// ---------------------------------------------------------------------------
__global__ void dlstm_pack(const float* __restrict__ Wis,
                           const float* __restrict__ bis,
                           const float* __restrict__ Wss,
                           const float* __restrict__ bss,
                           unsigned short* __restrict__ wisP,
                           unsigned short* __restrict__ w01P,
                           float* __restrict__ biasP) {
    int gid = blockIdx.x * blockDim.x + threadIdx.x;
    if (gid < 65536) {                       // W_is frags: [mt(32)][kc(4)][lane(32)][e(16)]
        int e  = gid & 15;
        int l  = (gid >> 4) & 31;
        int kc = (gid >> 9) & 3;
        int m  = gid >> 11;
        int Mp = m * 16 + (l & 15);                      // permuted row
        int Mo = (Mp & 3) * C_ + (Mp >> 2);              // orig = gate*128 + ch
        int K  = kc * 32 + ((l >> 4) * 16) + e;
        wisP[gid] = f2bf(Wis[Mo * C_ + K]);
    } else if (gid < 65536 + 131072) {       // [W0|W1] frags: [mt(32)][kc(8)][lane][e]
        int g  = gid - 65536;
        int e  = g & 15;
        int l  = (g >> 4) & 31;
        int kc = (g >> 9) & 7;
        int m  = g >> 12;
        int Mp = m * 16 + (l & 15);
        int Mo = (Mp & 3) * C_ + (Mp >> 2);
        int K  = kc * 32 + ((l >> 4) * 16) + e;
        // K < 128 -> W0 = W_ss[:,:,0] (multiplies h_up); else W1 = W_ss[:,:,1]
        float v = (K < C_) ? Wss[(Mo * C_ + K) * 2 + 0]
                           : Wss[(Mo * C_ + (K - C_)) * 2 + 1];
        w01P[g] = f2bf(v);
    } else if (gid < 65536 + 131072 + O_) {
        int o = gid - (65536 + 131072);      // orig order: gate*128 + ch
        biasP[o] = bis[o] + bss[o];
    }
}

// ---------------------------------------------------------------------------
// Main persistent kernel: one workgroup per batch.
// Dynamic LDS (304KB):
//   w01L   [32][8][32][16] bf16  (256KB)  packed [W0|W1] A-fragments
//   xbf[2] [32][128]       bf16  (16KB)   x_t B-matrix staging, double-buffered
//   hbf[2] [32][256]       bf16  (32KB)   [h_up;h] B-matrix staging, dbl-buf
// ---------------------------------------------------------------------------
__global__ __launch_bounds__(1024) void dlstm_main(
        const float* __restrict__ x,
        const unsigned short* __restrict__ wisP,
        const unsigned short* __restrict__ w01P,
        const float* __restrict__ biasP,
        float* __restrict__ out) {
    extern __shared__ char smem[];
    unsigned short* w01L = (unsigned short*)smem;                   // 131072 elems
    unsigned short* xb0  = (unsigned short*)(smem + 262144);        // 4096
    unsigned short* xb1  = (unsigned short*)(smem + 262144 + 8192); // 4096
    unsigned short* hb0  = (unsigned short*)(smem + 278528);        // 8192
    unsigned short* hb1  = (unsigned short*)(smem + 278528 + 16384);// 8192

    const int b    = blockIdx.x;
    const int tid  = (int)threadIdx.x;
    const int wid  = tid >> 5;          // 0..31: M-tile (16 permuted rows = 4 channels)
    const int lane = tid & 31;
    const int ln   = lane & 15;         // N (h-row) within tile
    const int hi   = lane >> 4;
    const int kh   = hi * 16;           // B-frag K offset for upper half-lanes

    // ---- Cooperative copy of [W0|W1] fragments into LDS (256KB) ----
    {
        const uint4* src = (const uint4*)w01P;
        uint4* dst = (uint4*)w01L;
#pragma unroll
        for (int j = 0; j < 16; ++j) dst[tid + j * 1024] = src[tid + j * 1024];
    }

    // ---- W_is A-fragments in registers (constant over t): 32 VGPRs ----
    v16bf aW[4];
    {
        const v16bf* wF = (const v16bf*)wisP;
#pragma unroll
        for (int kc = 0; kc < 4; ++kc) aW[kc] = wF[(wid * 4 + kc) * 32 + lane];
    }

    // ---- Per-lane cells: 2 channels x 2 N-tiles ----
    // channel ch = wid*4 + 2*hi + chl ; h-row n = nt*16 + ln
    const int ch0 = wid * 4 + 2 * hi;
    float cst[4] = {0.f, 0.f, 0.f, 0.f};        // [nt*2 + chl]
    float bsv[2][4];                            // [chl][gate]
#pragma unroll
    for (int chl = 0; chl < 2; ++chl)
#pragma unroll
        for (int gt = 0; gt < 4; ++gt)
            bsv[chl][gt] = biasP[gt * C_ + ch0 + chl];

    // ---- Init: zero both h staging buffers; stage x for t=0 into xb0 ----
#pragma unroll
    for (int j = 0; j < 8; ++j) { hb0[tid + j * 1024] = 0; hb1[tid + j * 1024] = 0; }
#pragma unroll
    for (int nt = 0; nt < 2; ++nt)
#pragma unroll
        for (int chl = 0; chl < 2; ++chl) {
            int n = nt * 16 + ln, c = ch0 + chl;
            unsigned w = (unsigned)(0 - n);
            float v = (w < W_) ? x[((b * C_ + c) * H_ + n) * W_ + (int)w] : 0.f;
            xb0[n * C_ + c] = f2bf(v);
        }
    __syncthreads();

    for (int t = 0; t < T_; ++t) {
        unsigned short* xc = (t & 1) ? xb1 : xb0;   // current staging
        unsigned short* xn = (t & 1) ? xb0 : xb1;   // next staging
        unsigned short* hc = (t & 1) ? hb1 : hb0;
        unsigned short* hn_ = (t & 1) ? hb0 : hb1;

        // ---- Fused GEMMs: acc = W_is@x_t + [W0|W1]@[h_up;h] ----
        v8f acc0 = {};   // N tile 0 (h rows 0..15)
        v8f acc1 = {};   // N tile 1 (h rows 16..31)
#pragma unroll
        for (int kc = 0; kc < 4; ++kc) {
            v16bf b0 = *(const v16bf*)(xc + ln * C_        + kc * 32 + kh);
            v16bf b1 = *(const v16bf*)(xc + (16 + ln) * C_ + kc * 32 + kh);
            acc0 = __builtin_amdgcn_wmma_f32_16x16x32_bf16(false, aW[kc], false, b0,
                                                           (short)0, acc0, false, false);
            acc1 = __builtin_amdgcn_wmma_f32_16x16x32_bf16(false, aW[kc], false, b1,
                                                           (short)0, acc1, false, false);
        }
#pragma unroll
        for (int kc = 0; kc < 8; ++kc) {
            v16bf aU = *(const v16bf*)(w01L + ((wid * 8 + kc) * 32 + lane) * 16);
            v16bf b0 = *(const v16bf*)(hc + ln * (2 * C_)        + kc * 32 + kh);
            v16bf b1 = *(const v16bf*)(hc + (16 + ln) * (2 * C_) + kc * 32 + kh);
            acc0 = __builtin_amdgcn_wmma_f32_16x16x32_bf16(false, aU, false, b0,
                                                           (short)0, acc0, false, false);
            acc1 = __builtin_amdgcn_wmma_f32_16x16x32_bf16(false, aU, false, b1,
                                                           (short)0, acc1, false, false);
        }

        // ---- Gate math directly on accumulators (no z round-trip) ----
        // acc element r: gate = r&3, chl = r>>2 ; channel = ch0 + chl
#pragma unroll
        for (int nt = 0; nt < 2; ++nt) {
            const v8f& a = nt ? acc1 : acc0;
            int n = nt * 16 + ln;
#pragma unroll
            for (int chl = 0; chl < 2; ++chl) {
                int c = ch0 + chl;
                float zo = a[chl * 4 + 0] + bsv[chl][0];
                float zf = a[chl * 4 + 1] + bsv[chl][1];
                float zi = a[chl * 4 + 2] + bsv[chl][2];
                float zg = a[chl * 4 + 3] + bsv[chl][3];
                float so = 1.f / (1.f + __expf(-zo));
                float sf = 1.f / (1.f + __expf(-zf));
                float si = 1.f / (1.f + __expf(-zi));
                float g  = tanhf(zg);
                float cn = sf * cst[nt * 2 + chl] + si * g;
                cst[nt * 2 + chl] = cn;
                float hv = so * tanhf(cn);

                // un-skewed output: out[b,c,n,w], w = t - n
                unsigned w = (unsigned)(t - n);
                if (w < W_) out[((b * C_ + c) * H_ + n) * W_ + (int)w] = hv;

                // stage h for t+1: k>=128 -> h ; k<128 -> h_up (shift +1 in H)
                unsigned short hbv = f2bf(hv);
                hn_[n * (2 * C_) + C_ + c] = hbv;
                if (n + 1 < H_) hn_[(n + 1) * (2 * C_) + c] = hbv;

                // stage x for t+1 (band-masked skew read)
                unsigned wn = (unsigned)(t + 1 - n);
                float xv = (wn < W_) ? x[((b * C_ + c) * H_ + n) * W_ + (int)wn] : 0.f;
                xn[n * C_ + c] = f2bf(xv);
            }
        }
        __syncthreads();
    }
}

// ---------------------------------------------------------------------------
// Launch: inputs (all f32): x, W_is, b_is, W_ss, b_ss
// ws: wisP bf16 @0 (128KB) | w01P bf16 @131072 (256KB) | bias f32 @393216
// ---------------------------------------------------------------------------
extern "C" void kernel_launch(void* const* d_in, const int* in_sizes, int n_in,
                              void* d_out, int out_size, void* d_ws, size_t ws_size,
                              hipStream_t stream) {
    (void)in_sizes; (void)n_in; (void)out_size; (void)ws_size;
    const float* x   = (const float*)d_in[0];
    const float* Wis = (const float*)d_in[1];
    const float* bis = (const float*)d_in[2];
    const float* Wss = (const float*)d_in[3];
    const float* bss = (const float*)d_in[4];

    char* ws = (char*)d_ws;
    unsigned short* wisP  = (unsigned short*)ws;                 // 65536  bf16
    unsigned short* w01P  = (unsigned short*)(ws + 131072);      // 131072 bf16
    float*          biasP = (float*)(ws + 131072 + 262144);      // 512    f32

    dlstm_pack<<<770, 256, 0, stream>>>(Wis, bis, Wss, bss, wisP, w01P, biasP);

    // one WG per batch; 304KB dynamic LDS (w01 frags + double-buffered x/h)
    dlstm_main<<<B_, 1024, 311296, stream>>>(x, wisP, w01P, biasP, (float*)d_out);
}
